// BagModel_4904852652359
// MI455X (gfx1250) — matrix operand order repaired and checked by reading, not compiled, not run on hardware
//
#include <hip/hip_runtime.h>
#include <hip/hip_bf16.h>

typedef __attribute__((ext_vector_type(16))) _Float16 v16h;
typedef __attribute__((ext_vector_type(8)))  float    v8f;

#define T_TOTAL 2000000
#define NBAGS   100000
#define DIN     64
#define DHID    128

// ---------------------------------------------------------------------------
// Kernel 1: fused instance MLP.  y[i] = relu(x[i]·W1 + b1)·W2  via f16 WMMA.
// One wave handles a 16-instance tile: 8 N-tiles x 2 independent K-step
// WMMAs (v_wmma_f32_16x16x32_f16), fp32 accumulate, b1 folded into C-seed.
// nt loop kept at unroll-2 so B fragments stay in LDS (not registers):
// the kernel is HBM-bound (512 MB of x / 23.3 TB/s ~ 22 us), so occupancy
// beats ILP depth here.
// ---------------------------------------------------------------------------
__global__ __launch_bounds__(256) void mil_wmma_kernel(
    const float* __restrict__ x,   // [T, 64]
    const float* __restrict__ W1,  // [64, 128] row-major (K x N)
    const float* __restrict__ b1,  // [128]
    const float* __restrict__ W2,  // [128]
    float* __restrict__ y,         // [T] out
    int ntiles)
{
    // B fragments pre-swizzled: [nt(8)][kh(2)][lane(32)][16 halves] = 16 KB.
    // Dense-B layout (ISA 7.12.5 scaled to K=32): lane<16 -> N=lane, K={0..7,16..23}
    // in halves; lane>=16 -> N=lane-16, K={8..15,24..31}; kh adds K+=32.
    __shared__ __align__(32) _Float16 ldsB[8192];
    __shared__ float ldsB1[DHID];
    __shared__ float ldsW2[DHID];

    const int t = threadIdx.x;
    for (int s = t; s < 8192; s += 256) {
        int h    = s & 15;
        int lane = (s >> 4) & 31;
        int kh   = (s >> 9) & 1;
        int nt   = s >> 10;
        int N = nt * 16 + (lane & 15);
        int K = kh * 32 + ((h >> 3) << 4) + ((lane >> 4) << 3) + (h & 7);
        ldsB[s] = (_Float16)W1[K * DHID + N];
    }
    if (t < DHID) { ldsB1[t] = b1[t]; ldsW2[t] = W2[t]; }
    __syncthreads();

    const int lane = t & 31;
    const int ln   = lane & 15;        // position within 16-lane half
    const int hi   = lane >> 4;        // which half (0/1)
    const int waveId = blockIdx.x * (blockDim.x >> 5) + (t >> 5);
    const int nWaves = gridDim.x * (blockDim.x >> 5);

    for (int tile = waveId; tile < ntiles; tile += nWaves) {
        // Prefetch next tile's row for this lane (global_prefetch_b8).
        {
            int nxt = tile + nWaves;
            if (nxt < ntiles)
                __builtin_prefetch(x + (size_t)(nxt * 16 + ln) * DIN, 0, 3);
        }

        // ---- load one x row per lane-pair, build A fragments (f32 -> f16) ----
        // A layout (ISA 16-bit A 16x32): lane<16: M=lane, K in {0..7,16..23};
        // lane>=16: M=lane-16, K in {8..15,24..31}.  Fragment a1 adds K+=32.
        const int row = tile * 16 + ln;
        const float4* xp = (const float4*)(x + (size_t)row * DIN);
        const int c4 = hi * 2;                 // float4 index of first chunk
        float4 f0a = xp[c4 + 0],  f0b = xp[c4 + 1];    // K base + {0..7 | 8..15}
        float4 f1a = xp[c4 + 4],  f1b = xp[c4 + 5];    // +16
        float4 f2a = xp[c4 + 8],  f2b = xp[c4 + 9];    // +32
        float4 f3a = xp[c4 + 12], f3b = xp[c4 + 13];   // +48

        v16h a0, a1;
        a0[0]  = (_Float16)f0a.x; a0[1]  = (_Float16)f0a.y;
        a0[2]  = (_Float16)f0a.z; a0[3]  = (_Float16)f0a.w;
        a0[4]  = (_Float16)f0b.x; a0[5]  = (_Float16)f0b.y;
        a0[6]  = (_Float16)f0b.z; a0[7]  = (_Float16)f0b.w;
        a0[8]  = (_Float16)f1a.x; a0[9]  = (_Float16)f1a.y;
        a0[10] = (_Float16)f1a.z; a0[11] = (_Float16)f1a.w;
        a0[12] = (_Float16)f1b.x; a0[13] = (_Float16)f1b.y;
        a0[14] = (_Float16)f1b.z; a0[15] = (_Float16)f1b.w;
        a1[0]  = (_Float16)f2a.x; a1[1]  = (_Float16)f2a.y;
        a1[2]  = (_Float16)f2a.z; a1[3]  = (_Float16)f2a.w;
        a1[4]  = (_Float16)f2b.x; a1[5]  = (_Float16)f2b.y;
        a1[6]  = (_Float16)f2b.z; a1[7]  = (_Float16)f2b.w;
        a1[8]  = (_Float16)f3a.x; a1[9]  = (_Float16)f3a.y;
        a1[10] = (_Float16)f3a.z; a1[11] = (_Float16)f3a.w;
        a1[12] = (_Float16)f3b.x; a1[13] = (_Float16)f3b.y;
        a1[14] = (_Float16)f3b.z; a1[15] = (_Float16)f3b.w;

        float acc[8];
#pragma unroll
        for (int j = 0; j < 8; ++j) acc[j] = 0.0f;

#pragma unroll 2
        for (int nt = 0; nt < 8; ++nt) {
            const float b1v = ldsB1[nt * 16 + ln];
            const float w2v = ldsW2[nt * 16 + ln];

            // Two INDEPENDENT accumulators (no WMMA->WMMA D->C RAW chain):
            // c0 seeded with the b1 broadcast, c1 seeded with zero.
            v8f c0, c1;
#pragma unroll
            for (int j = 0; j < 8; ++j) { c0[j] = b1v; c1[j] = 0.0f; }

            const v16h bf0 = *reinterpret_cast<const v16h*>(
                &ldsB[((nt * 2 + 0) * 32 + lane) * 16]);
            const v16h bf1 = *reinterpret_cast<const v16h*>(
                &ldsB[((nt * 2 + 1) * 32 + lane) * 16]);

            c0 = __builtin_amdgcn_wmma_f32_16x16x32_f16(
                     false, a0, false, bf0, (short)0, c0, false, false);
            c1 = __builtin_amdgcn_wmma_f32_16x16x32_f16(
                     false, a1, false, bf1, (short)0, c1, false, false);

            // C layout: VGPR j on lane L = h[M = j + 8*hi][N = nt*16 + ln]
#pragma unroll
            for (int j = 0; j < 8; ++j)
                acc[j] += fmaxf(c0[j] + c1[j], 0.0f) * w2v;
        }

        // reduce each acc[j] across the 16-lane half (columns N)
#pragma unroll
        for (int m = 1; m < 16; m <<= 1) {
#pragma unroll
            for (int j = 0; j < 8; ++j)
                acc[j] += __shfl_xor(acc[j], m, 32);
        }

        // lane (hi*16 + j) writes y for row tile*16 + hi*8 + j
        float* yp = y + tile * 16 + hi * 8;
#pragma unroll
        for (int j = 0; j < 8; ++j)
            if (ln == j) yp[j] = acc[j];
    }
}

// ---------------------------------------------------------------------------
// Prefix-sum of n_instances (needed to reproduce the reference's
// total_repeat_length=T truncation semantics exactly).
// ---------------------------------------------------------------------------
__global__ __launch_bounds__(1024) void scan_block_kernel(
    const int* __restrict__ n, unsigned* __restrict__ incl,
    unsigned* __restrict__ blockSums, int B)
{
    __shared__ unsigned s[1024];
    const int i = blockIdx.x * 1024 + threadIdx.x;
    unsigned v = (i < B) ? (unsigned)n[i] : 0u;
    s[threadIdx.x] = v;
    __syncthreads();
    for (int off = 1; off < 1024; off <<= 1) {
        unsigned add = (threadIdx.x >= (unsigned)off) ? s[threadIdx.x - off] : 0u;
        __syncthreads();
        s[threadIdx.x] += add;
        __syncthreads();
    }
    if (i < B) incl[i] = s[threadIdx.x];          // within-block inclusive scan
    if (threadIdx.x == 1023) blockSums[blockIdx.x] = s[1023];
}

__global__ __launch_bounds__(128) void scan_sums_kernel(
    const unsigned* __restrict__ blockSums, unsigned* __restrict__ blockOffs,
    int nb)
{
    __shared__ unsigned s[128];
    const int t = threadIdx.x;
    unsigned v = (t < nb) ? blockSums[t] : 0u;
    s[t] = v;
    __syncthreads();
    for (int off = 1; off < 128; off <<= 1) {
        unsigned add = (t >= off) ? s[t - off] : 0u;
        __syncthreads();
        s[t] += add;
        __syncthreads();
    }
    if (t < nb) blockOffs[t] = s[t] - v;          // exclusive block offsets
}

// ---------------------------------------------------------------------------
// Ragged segment mean with truncation at T, + b2.
// ---------------------------------------------------------------------------
__global__ __launch_bounds__(256) void bag_mean_kernel(
    const float* __restrict__ y, const int* __restrict__ n,
    const unsigned* __restrict__ incl, const unsigned* __restrict__ blockOffs,
    const float* __restrict__ b2, float* __restrict__ out, int B)
{
    const int b = blockIdx.x * blockDim.x + threadIdx.x;
    if (b >= B) return;
    const unsigned gi = incl[b] + blockOffs[b >> 10];   // global inclusive sum
    const unsigned nb = (unsigned)n[b];
    const unsigned T  = (unsigned)T_TOTAL;
    unsigned e = gi < T ? gi : T;
    unsigned s = (gi - nb) < T ? (gi - nb) : T;
    float sum = 0.0f;
    for (unsigned i = s; i < e; ++i) sum += y[i];
    const unsigned c = e - s;
    out[b] = (c ? sum / (float)c : 0.0f) + b2[0];
}

// ---------------------------------------------------------------------------
extern "C" void kernel_launch(void* const* d_in, const int* in_sizes, int n_in,
                              void* d_out, int out_size, void* d_ws, size_t ws_size,
                              hipStream_t stream) {
    const float* x  = (const float*)d_in[0];
    const int*   n  = (const int*)  d_in[1];
    const float* W1 = (const float*)d_in[2];
    const float* b1 = (const float*)d_in[3];
    const float* W2 = (const float*)d_in[4];
    const float* b2 = (const float*)d_in[5];
    float* out = (float*)d_out;

    char* ws = (char*)d_ws;
    float*    y         = (float*)ws;                                   // T floats
    unsigned* incl      = (unsigned*)(ws + sizeof(float) * T_TOTAL);    // B
    unsigned* blockSums = incl + NBAGS;                                 // <=128
    unsigned* blockOffs = blockSums + 128;                              // <=128

    const int ntiles = T_TOTAL / 16;            // 125000, exact
    mil_wmma_kernel<<<512, 256, 0, stream>>>(x, W1, b1, W2, y, ntiles);

    const int nb = (NBAGS + 1023) / 1024;       // 98 blocks
    scan_block_kernel<<<nb, 1024, 0, stream>>>(n, incl, blockSums, NBAGS);
    scan_sums_kernel<<<1, 128, 0, stream>>>(blockSums, blockOffs, nb);
    bag_mean_kernel<<<(NBAGS + 255) / 256, 256, 0, stream>>>(
        y, n, incl, blockOffs, b2, out, NBAGS);
}
